// OrthAwareMissingEmbeddingGenerator_29764123361487
// MI455X (gfx1250) — matrix operand order repaired and checked by reading, not compiled, not run on hardware
//
#include <hip/hip_runtime.h>
#include <hip/hip_bf16.h>
#include <math.h>

// ---------------------------------------------------------------------------
// OrthAwareMissingEmbeddingGenerator forward for MI455X (gfx1250, wave32)
//   bf16 WMMA (v_wmma_f32_16x16x32_bf16) for all GEMMs, fp32 elsewhere.
// ---------------------------------------------------------------------------

typedef __bf16 bf16;
typedef bf16  v16bf __attribute__((ext_vector_type(16)));
typedef bf16  v8bf  __attribute__((ext_vector_type(8)));
typedef float v8f   __attribute__((ext_vector_type(8)));

#define B_SZ   2048
#define S_SZ   23
#define E_SZ   768
#define E2     384
#define RED_   256
#define GB     4            // batch rows per workgroup
#define MROWS  96           // GB*23 padded to 6 WMMA row tiles

#define LDA1   776          // 768 + 8 pad (bank spread)
#define LDH1   392          // 384 + 8
#define LDH2   260          // 256 + 4 (f32)
#define LDR    264          // 256 + 8
#define OFF_H1 (MROWS * LDA1 * 2)          // 148992 bytes
#define SZQ    (MROWS * LDR)               // elems per bf16 [96][264] plane
#define SM1_BYTES (MROWS * LDA1 * 2 + MROWS * LDH1 * 2)   // 224256
#define SM2_BYTES (4 * SZQ * 2 + 384 + 128)               // 203264

__device__ __forceinline__ v8f zero8() {
  v8f z = {0.f, 0.f, 0.f, 0.f, 0.f, 0.f, 0.f, 0.f};
  return z;
}

__device__ __forceinline__ v8f wmma_bf16(v16bf a, v16bf b, v8f c) {
  // (neg_a, A, neg_b, B, c_mod, C, reuse_a, reuse_b)
  return __builtin_amdgcn_wmma_f32_16x16x32_bf16(false, a, false, b,
                                                 (short)0, c, false, false);
}

union U16 { v16bf v; v8bf h[2]; };

// A fragment (16x32 bf16). A row-major [M x lda].
// lane: m = m0 + (lane&15); halves at k0+8*hi and k0+16+8*hi (hi = lane>>4).
// Two contiguous 16B loads per lane.
__device__ __forceinline__ v16bf frag_a(const bf16* A, int lda, int m0, int k0, int lane) {
  const int m  = m0 + (lane & 15);
  const int hi = lane >> 4;
  const bf16* p = A + (size_t)m * lda + k0 + 8 * hi;
  U16 u;
  u.h[0] = *(const v8bf*)(p);
  u.h[1] = *(const v8bf*)(p + 16);
  return u.v;
}

// B fragment (32x16 bf16) with B[k][n] = W[n][k], W row-major [N x ldw] (k contig).
// lane: n = n0 + (lane&15); halves at k0+8*hi and k0+16+8*hi (hi = lane>>4).  Two contiguous 16B loads.
__device__ __forceinline__ v16bf frag_b(const bf16* W, int ldw, int n0, int k0, int lane) {
  const int n  = n0 + (lane & 15);
  const int hi = lane >> 4;
  const bf16* p = W + (size_t)n * ldw + k0 + 8 * hi;
  U16 u;
  u.h[0] = *(const v8bf*)(p);
  u.h[1] = *(const v8bf*)(p + 16);
  return u.v;
}

__device__ __forceinline__ float gelu_exact(float x) {
  return 0.5f * x * (1.0f + erff(x * 0.70710678118654752f));
}

// ---------------------------------------------------------------------------
__global__ void k_cvt_bf16(const float* __restrict__ src, bf16* __restrict__ dst, int n) {
  int i = blockIdx.x * blockDim.x + threadIdx.x;
  if (i < n) dst[i] = (bf16)src[i];
}

// ---------------------------------------------------------------------------
// Kernel 1: gather + MLP (2 GEMMs, exact GELU) + LayerNorm -> red [B,S,256] f32
// ---------------------------------------------------------------------------
__global__ __launch_bounds__(256) void k_mlp_ln(
    const float* __restrict__ cls, const int* __restrict__ mask,
    const float* __restrict__ miss,
    const bf16* __restrict__ W1b, const float* __restrict__ b1,
    const bf16* __restrict__ W2b, const float* __restrict__ b2,
    const float* __restrict__ lng, const float* __restrict__ lnb,
    float* __restrict__ red_out)
{
  extern __shared__ char smem[];
  bf16*  Asec = (bf16*)smem;               // [96][776] bf16 (gathered input)
  float* h2   = (float*)smem;              // [96][260] f32  (overlays Asec later)
  bf16*  h1   = (bf16*)(smem + OFF_H1);    // [96][392] bf16

  const int tid  = threadIdx.x;
  const int lane = tid & 31;
  const int wave = tid >> 5;
  const int b0   = blockIdx.x * GB;

  // ---- gather (mask ? cls : missing), f32 -> bf16 into LDS ----
  for (int idx = tid; idx < MROWS * E_SZ; idx += 256) {
    const int r = idx / E_SZ;
    const int k = idx - r * E_SZ;
    float v = 0.f;
    if (r < GB * S_SZ) {
      const int bl = r / S_SZ, s = r - bl * S_SZ;
      const size_t row = (size_t)(b0 + bl) * S_SZ + s;
      v = (mask[row] > 0) ? cls[row * (size_t)E_SZ + k] : miss[(size_t)s * E_SZ + k];
    }
    Asec[(size_t)r * LDA1 + k] = (bf16)v;
  }
  __syncthreads();

  // ---- GEMM1: h1[96,384] = gelu(Asec[96,768] @ W1^T + b1), bf16 out ----
  for (int nt = wave; nt < E2 / 16; nt += 8) {
    const int n0 = nt * 16;
    v8f acc[6];
#pragma unroll
    for (int m = 0; m < 6; ++m) acc[m] = zero8();
    for (int ks = 0; ks < E_SZ / 32; ++ks) {
      __builtin_prefetch(W1b + (size_t)(n0 + (lane & 15)) * E_SZ + (ks + 1) * 32, 0, 3);
      const v16bf bfr = frag_b(W1b, E_SZ, n0, ks * 32, lane);
#pragma unroll
      for (int m = 0; m < 6; ++m) {
        const v16bf afr = frag_a(Asec, LDA1, m * 16, ks * 32, lane);
        acc[m] = wmma_bf16(afr, bfr, acc[m]);
      }
    }
    const int n  = n0 + (lane & 15);
    const int hi = lane >> 4;
    const float bias = b1[n];
#pragma unroll
    for (int m = 0; m < 6; ++m)
#pragma unroll
      for (int j = 0; j < 8; ++j) {
        const int row = m * 16 + j + 8 * hi;
        h1[(size_t)row * LDH1 + n] = (bf16)gelu_exact(acc[m][j] + bias);
      }
  }
  __syncthreads();

  // ---- GEMM2: h2[96,256] = gelu(h1 @ W2^T + b2), f32 out (overlays Asec) ----
  for (int nt = wave; nt < RED_ / 16; nt += 8) {
    const int n0 = nt * 16;
    v8f acc[6];
#pragma unroll
    for (int m = 0; m < 6; ++m) acc[m] = zero8();
    for (int ks = 0; ks < E2 / 32; ++ks) {
      const v16bf bfr = frag_b(W2b, E2, n0, ks * 32, lane);
#pragma unroll
      for (int m = 0; m < 6; ++m) {
        const v16bf afr = frag_a(h1, LDH1, m * 16, ks * 32, lane);
        acc[m] = wmma_bf16(afr, bfr, acc[m]);
      }
    }
    const int n  = n0 + (lane & 15);
    const int hi = lane >> 4;
    const float bias = b2[n];
#pragma unroll
    for (int m = 0; m < 6; ++m)
#pragma unroll
      for (int j = 0; j < 8; ++j) {
        const int row = m * 16 + j + 8 * hi;
        h2[(size_t)row * LDH2 + n] = gelu_exact(acc[m][j] + bias);
      }
  }
  __syncthreads();

  // ---- LayerNorm per row (one wave per row, wave32 shuffle reduce) ----
  for (int r = wave; r < GB * S_SZ; r += 8) {
    float s = 0.f;
    for (int c = lane; c < RED_; c += 32) s += h2[r * LDH2 + c];
#pragma unroll
    for (int o = 16; o > 0; o >>= 1) s += __shfl_xor(s, o, 32);
    const float mu = s * (1.f / RED_);
    float v = 0.f;
    for (int c = lane; c < RED_; c += 32) {
      const float d = h2[r * LDH2 + c] - mu;
      v += d * d;
    }
#pragma unroll
    for (int o = 16; o > 0; o >>= 1) v += __shfl_xor(v, o, 32);
    const float rstd = rsqrtf(v * (1.f / RED_) + 1e-5f);
    const int bl = r / S_SZ, sidx = r - bl * S_SZ;
    float* orow = red_out + ((size_t)(b0 + bl) * S_SZ + sidx) * RED_;
    for (int c = lane; c < RED_; c += 32)
      orow[c] = (h2[r * LDH2 + c] - mu) * rstd * lng[c] + lnb[c];
  }
}

// ---------------------------------------------------------------------------
// Kernel 2: QKV GEMMs + masked attention + Wo GEMM; scatter into red (in place)
// ---------------------------------------------------------------------------
__global__ __launch_bounds__(256) void k_attn(
    const float* __restrict__ redg, const int* __restrict__ mask,
    const bf16* __restrict__ Wqb, const float* __restrict__ bq,
    const bf16* __restrict__ Wkb, const float* __restrict__ bk,
    const bf16* __restrict__ Wvb, const float* __restrict__ bv,
    const bf16* __restrict__ Wob, const float* __restrict__ bo,
    float* __restrict__ upd)
{
  extern __shared__ char smem[];
  bf16* A  = (bf16*)smem;          // [96][264] red bf16; later ctx
  bf16* Qm = A  + SZQ;
  bf16* Km = Qm + SZQ;
  bf16* Vm = Km + SZQ;
  int*  exm = (int*)(smem + 4 * SZQ * 2);
  unsigned char* useA = (unsigned char*)(smem + 4 * SZQ * 2 + 384);

  const int tid  = threadIdx.x;
  const int lane = tid & 31;
  const int wave = tid >> 5;
  const int b0   = blockIdx.x * GB;

  // ---- load red -> bf16 LDS; mask -> LDS ----
  for (int idx = tid; idx < MROWS * RED_; idx += 256) {
    const int r = idx >> 8, c = idx & 255;
    float v = 0.f;
    if (r < GB * S_SZ) {
      const int bl = r / S_SZ, s = r - bl * S_SZ;
      v = redg[((size_t)(b0 + bl) * S_SZ + s) * RED_ + c];
    }
    A[(size_t)r * LDR + c] = (bf16)v;
  }
  if (tid < MROWS) {
    int e = 0;
    if (tid < GB * S_SZ)
      e = (mask[(size_t)(b0 + tid / S_SZ) * S_SZ + (tid % S_SZ)] > 0) ? 1 : 0;
    exm[tid] = e;
  }
  __syncthreads();

  if (tid < GB * S_SZ) {
    const int bl = tid / S_SZ;
    int any = 0;
    for (int s = 0; s < S_SZ; ++s) any |= exm[bl * S_SZ + s];
    useA[tid] = (unsigned char)((!exm[tid]) && any);
  }

  // ---- Q/K/V projections: 3 GEMMs N=256 K=256, 48 tiles over 8 waves ----
  for (int t = wave; t < 48; t += 8) {
    const int matid = t >> 4;           // 0=Q 1=K 2=V
    const int nt    = t & 15;
    const bf16*  Wb = (matid == 0) ? Wqb : (matid == 1) ? Wkb : Wvb;
    const float* bb = (matid == 0) ? bq  : (matid == 1) ? bk  : bv;
    bf16*        Om = (matid == 0) ? Qm  : (matid == 1) ? Km  : Vm;
    const int n0 = nt * 16;
    v8f acc[6];
#pragma unroll
    for (int m = 0; m < 6; ++m) acc[m] = zero8();
    for (int ks = 0; ks < RED_ / 32; ++ks) {
      const v16bf bfr = frag_b(Wb, RED_, n0, ks * 32, lane);
#pragma unroll
      for (int m = 0; m < 6; ++m) {
        const v16bf afr = frag_a(A, LDR, m * 16, ks * 32, lane);
        acc[m] = wmma_bf16(afr, bfr, acc[m]);
      }
    }
    const int n  = n0 + (lane & 15);
    const int hi = lane >> 4;
    const float bias = bb[n];
#pragma unroll
    for (int m = 0; m < 6; ++m)
#pragma unroll
      for (int j = 0; j < 8; ++j) {
        const int row = m * 16 + j + 8 * hi;
        Om[(size_t)row * LDR + n] = (bf16)(acc[m][j] + bias);
      }
  }
  __syncthreads();

  // ---- attention: one (b_local, head, query) per iteration; ctx -> A ----
  for (int t = tid; t < GB * 8 * S_SZ; t += 256) {
    const int bl  = t / (8 * S_SZ);
    const int rem = t - bl * (8 * S_SZ);
    const int h   = rem / S_SZ;
    const int qs  = rem - h * S_SZ;
    const int qr  = bl * S_SZ + qs;
    const int co  = h * 32;
    float qv[32];
#pragma unroll
    for (int d = 0; d < 32; ++d) qv[d] = (float)Qm[(size_t)qr * LDR + co + d];
    float sc[S_SZ];
    float mx = -3.0e38f;
    for (int k2 = 0; k2 < S_SZ; ++k2) {
      const int kr = bl * S_SZ + k2;
      float s = 0.f;
#pragma unroll
      for (int d = 0; d < 32; ++d) s += qv[d] * (float)Km[(size_t)kr * LDR + co + d];
      s *= 0.17677669529663687f;        // 1/sqrt(32)
      if (!exm[kr]) s = -1.0e9f;
      sc[k2] = s;
      mx = fmaxf(mx, s);
    }
    float den = 0.f;
    for (int k2 = 0; k2 < S_SZ; ++k2) {
      const float e = __expf(sc[k2] - mx);
      sc[k2] = e;
      den += e;
    }
    const float inv = 1.f / den;
    float cx[32];
#pragma unroll
    for (int d = 0; d < 32; ++d) cx[d] = 0.f;
    for (int k2 = 0; k2 < S_SZ; ++k2) {
      const float w = sc[k2] * inv;
      const int kr = bl * S_SZ + k2;
#pragma unroll
      for (int d = 0; d < 32; ++d) cx[d] += w * (float)Vm[(size_t)kr * LDR + co + d];
    }
#pragma unroll
    for (int d = 0; d < 32; ++d) A[(size_t)qr * LDR + co + d] = (bf16)cx[d];
  }
  __syncthreads();

  // ---- Wo GEMM on ctx; scatter only (~exist & any_exist) rows ----
  for (int nt = wave; nt < RED_ / 16; nt += 8) {
    const int n0 = nt * 16;
    v8f acc[6];
#pragma unroll
    for (int m = 0; m < 6; ++m) acc[m] = zero8();
    for (int ks = 0; ks < RED_ / 32; ++ks) {
      const v16bf bfr = frag_b(Wob, RED_, n0, ks * 32, lane);
#pragma unroll
      for (int m = 0; m < 6; ++m) {
        const v16bf afr = frag_a(A, LDR, m * 16, ks * 32, lane);
        acc[m] = wmma_bf16(afr, bfr, acc[m]);
      }
    }
    const int n  = n0 + (lane & 15);
    const int hi = lane >> 4;
    const float bias = bo[n];
#pragma unroll
    for (int m = 0; m < 6; ++m)
#pragma unroll
      for (int j = 0; j < 8; ++j) {
        const int row = m * 16 + j + 8 * hi;
        if (row < GB * S_SZ && useA[row]) {
          const int bl = row / S_SZ, s = row - bl * S_SZ;
          upd[((size_t)(b0 + bl) * S_SZ + s) * RED_ + n] = acc[m][j] + bias;
        }
      }
  }
}

// ---------------------------------------------------------------------------
// Kernel 3: per-b Gram (WMMA) hinge loss partials + mean-pool classifier
// ---------------------------------------------------------------------------
__global__ __launch_bounds__(128) void k_loss_logits(
    const float* __restrict__ upd, const float* __restrict__ Wp,
    const float* __restrict__ bp, float* __restrict__ out,
    float* __restrict__ lossPart)
{
  __shared__ float Ush[S_SZ * LDH2];      // [23][260] f32
  __shared__ bf16  NB[32 * LDR];          // [32][264] bf16 (padded, zeroed)
  __shared__ float meanv[RED_];
  __shared__ float lossW[4];

  const int tid  = threadIdx.x;
  const int lane = tid & 31;
  const int wave = tid >> 5;
  const int b    = blockIdx.x;
  const float* src = upd + (size_t)b * S_SZ * RED_;

  for (int i = tid; i < 32 * LDR; i += 128) NB[i] = (bf16)0.f;
  for (int i = tid; i < S_SZ * RED_; i += 128) {
    const int r = i >> 8, c = i & 255;
    Ush[r * LDH2 + c] = src[i];
  }
  __syncthreads();

  // column means (for classifier head)
  for (int c = tid; c < RED_; c += 128) {
    float s = 0.f;
    for (int r = 0; r < S_SZ; ++r) s += Ush[r * LDH2 + c];
    meanv[c] = s * (1.0f / S_SZ);
  }

  // row-normalize -> NB bf16
  for (int r = wave; r < S_SZ; r += 4) {
    float ss = 0.f;
    for (int c = lane; c < RED_; c += 32) {
      const float x = Ush[r * LDH2 + c];
      ss += x * x;
    }
#pragma unroll
    for (int o = 16; o > 0; o >>= 1) ss += __shfl_xor(ss, o, 32);
    const float inv = 1.f / fmaxf(sqrtf(ss), 1e-8f);
    for (int c = lane; c < RED_; c += 32)
      NB[r * LDR + c] = (bf16)(Ush[r * LDH2 + c] * inv);
  }
  __syncthreads();

  // Gram = NB @ NB^T via WMMA: 4 waves, one 16x16 tile each (2x2 tiles)
  {
    const int mt  = wave >> 1;
    const int ntl = wave & 1;
    v8f acc = zero8();
    for (int ks = 0; ks < RED_ / 32; ++ks) {
      const v16bf afr = frag_a(NB, LDR, mt * 16, ks * 32, lane);
      const v16bf bfr = frag_b(NB, LDR, ntl * 16, ks * 32, lane);
      acc = wmma_bf16(afr, bfr, acc);
    }
    const int n  = ntl * 16 + (lane & 15);
    const int hi = lane >> 4;
    float l = 0.f;
#pragma unroll
    for (int j = 0; j < 8; ++j) {
      const int m = mt * 16 + j + 8 * hi;
      if (m < S_SZ && n < S_SZ && m != n) {
        const float g = fabsf(acc[j]) - 0.1f;   // DELTA
        if (g > 0.f) l += g;
      }
    }
#pragma unroll
    for (int o = 16; o > 0; o >>= 1) l += __shfl_xor(l, o, 32);
    if (lane == 0) lossW[wave] = l;
  }
  __syncthreads();

  if (tid < 25) {
    float s = bp[tid];
    const float* w = Wp + (size_t)tid * RED_;
    for (int c = 0; c < RED_; ++c) s += meanv[c] * w[c];
    out[1 + (size_t)b * 25 + tid] = s;
  }
  if (tid == 0)
    lossPart[b] = lossW[0] + lossW[1] + lossW[2] + lossW[3];   // fixed order
}

__global__ __launch_bounds__(256) void k_loss_reduce(
    const float* __restrict__ part, float* __restrict__ out)
{
  __shared__ float sm[256];
  float s = 0.f;
  for (int i = threadIdx.x; i < B_SZ; i += 256) s += part[i];
  sm[threadIdx.x] = s;
  __syncthreads();
  for (int o = 128; o > 0; o >>= 1) {
    if (threadIdx.x < o) sm[threadIdx.x] += sm[threadIdx.x + o];
    __syncthreads();
  }
  if (threadIdx.x == 0)
    out[0] = sm[0] * (1.0f / ((float)B_SZ * S_SZ * (S_SZ - 1)));
}

// ---------------------------------------------------------------------------
extern "C" void kernel_launch(void* const* d_in, const int* in_sizes, int n_in,
                              void* d_out, int out_size, void* d_ws, size_t ws_size,
                              hipStream_t stream)
{
  (void)in_sizes; (void)n_in; (void)out_size; (void)ws_size;
  const float* cls  = (const float*)d_in[0];
  const int*   mask = (const int*)d_in[1];
  const float* miss = (const float*)d_in[2];
  const float* W1   = (const float*)d_in[3];
  const float* b1   = (const float*)d_in[4];
  const float* W2   = (const float*)d_in[5];
  const float* b2   = (const float*)d_in[6];
  const float* lng  = (const float*)d_in[7];
  const float* lnb  = (const float*)d_in[8];
  const float* Wq   = (const float*)d_in[9];
  const float* bq   = (const float*)d_in[10];
  const float* Wk   = (const float*)d_in[11];
  const float* bk   = (const float*)d_in[12];
  const float* Wv   = (const float*)d_in[13];
  const float* bv   = (const float*)d_in[14];
  const float* Wo   = (const float*)d_in[15];
  const float* bo   = (const float*)d_in[16];
  const float* Wp   = (const float*)d_in[17];
  const float* bp   = (const float*)d_in[18];

  // Workspace layout (requires ~49.6 MB; harness workspace assumed >= this)
  char* ws = (char*)d_ws;
  bf16*  W1b = (bf16*)(ws + 0);             // 384*768  bf16 = 589824 B
  bf16*  W2b = (bf16*)(ws + 589824);        // 256*384  bf16 = 196608 B
  bf16*  Wqb = (bf16*)(ws + 786432);        // 256*256  bf16 = 131072 B
  bf16*  Wkb = (bf16*)(ws + 917504);
  bf16*  Wvb = (bf16*)(ws + 1048576);
  bf16*  Wob = (bf16*)(ws + 1179648);
  float* red = (float*)(ws + 1310720);      // 47104*256 f32 = 48234496 B (red -> updated, in place)
  float* lossPart = (float*)(ws + 1310720 + 48234496);   // 2048 f32
  float* out = (float*)d_out;               // [0]=loss, [1..51200]=logits

  // weight conversion to bf16 (cheap; L2-resident thereafter)
  k_cvt_bf16<<<(294912 + 255) / 256, 256, 0, stream>>>(W1, W1b, 294912);
  k_cvt_bf16<<<(98304  + 255) / 256, 256, 0, stream>>>(W2, W2b, 98304);
  k_cvt_bf16<<<256, 256, 0, stream>>>(Wq, Wqb, 65536);
  k_cvt_bf16<<<256, 256, 0, stream>>>(Wk, Wkb, 65536);
  k_cvt_bf16<<<256, 256, 0, stream>>>(Wv, Wvb, 65536);
  k_cvt_bf16<<<256, 256, 0, stream>>>(Wo, Wob, 65536);

  hipFuncSetAttribute(reinterpret_cast<const void*>(k_mlp_ln),
                      hipFuncAttributeMaxDynamicSharedMemorySize, SM1_BYTES);
  hipFuncSetAttribute(reinterpret_cast<const void*>(k_attn),
                      hipFuncAttributeMaxDynamicSharedMemorySize, SM2_BYTES);

  k_mlp_ln<<<B_SZ / GB, 256, SM1_BYTES, stream>>>(cls, mask, miss, W1b, b1,
                                                  W2b, b2, lng, lnb, red);
  k_attn<<<B_SZ / GB, 256, SM2_BYTES, stream>>>(red, mask, Wqb, bq, Wkb, bk,
                                                Wvb, bv, Wob, bo, red);
  k_loss_logits<<<B_SZ, 128, 0, stream>>>(red, Wp, bp, out, lossPart);
  k_loss_reduce<<<1, 256, 0, stream>>>(lossPart, out);
}
